// SlotAttention_6605659701496
// MI455X (gfx1250) — matrix operand order repaired and checked by reading, not compiled, not run on hardware
//
#include <hip/hip_runtime.h>

typedef __attribute__((ext_vector_type(16))) _Float16 v16h;
typedef __attribute__((ext_vector_type(8)))  _Float16 v8h;
typedef __attribute__((ext_vector_type(8)))  float    v8f;

#define BB 32
#define NN 16384
#define SS 7
#define DD 64
#define HH 128
#define MM (BB*NN)          // 524288
#define BS (BB*SS)          // 224
#define BSN (BB*SS*NN)      // 3670016
#define SNP (SS*NN)         // 114688 per-batch C elems
#define EPS8 1e-8f
#define LNE 1e-5f

__device__ __forceinline__ float wred(float v){
  #pragma unroll
  for (int o=16;o;o>>=1) v += __shfl_xor(v,o,32);
  return v;
}

// ---------------- slots init ----------------
__global__ __launch_bounds__(256) void slots_init(const float* __restrict__ mu,
    const float* __restrict__ sg, const float* __restrict__ noise, float* __restrict__ slots){
  int i = blockIdx.x*256 + threadIdx.x;
  if (i >= BS*DD) return;
  int d = i & 63;
  slots[i] = mu[d] + (fabsf(sg[d]) + EPS8) * noise[i];
}

// ---------------- LayerNorm(inputs) -> x_f16, a-logits ----------------
__global__ __launch_bounds__(256) void ln_a_kernel(const float* __restrict__ in,
    const float* __restrict__ g, const float* __restrict__ bb,
    const float* __restrict__ wi_w, const float* __restrict__ wi_b,
    _Float16* __restrict__ x16, float* __restrict__ alog){
  int wave = threadIdx.x >> 5, lane = threadIdx.x & 31;
  size_t row = (size_t)blockIdx.x*8 + wave;
  const float* p = in + row*64;
  float x0 = p[lane], x1 = p[lane+32];
  float mean = wred(x0+x1) * (1.f/64.f);
  float d0 = x0-mean, d1 = x1-mean;
  float rstd = rsqrtf(wred(d0*d0+d1*d1)*(1.f/64.f) + LNE);
  float y0 = d0*rstd*g[lane]    + bb[lane];
  float y1 = d1*rstd*g[lane+32] + bb[lane+32];
  x16[row*64+lane]    = (_Float16)y0;
  x16[row*64+lane+32] = (_Float16)y1;
  float dot = wred(y0*wi_w[lane] + y1*wi_w[lane+32]);
  if (lane==0) alog[row] = dot + wi_b[0];
}

// ---------------- softmax over N, *S, -> log_a ----------------
__global__ __launch_bounds__(1024) void a_softmax(float* __restrict__ a, float* __restrict__ log_a){
  __shared__ float red[1024];
  int b = blockIdx.x, t = threadIdx.x;
  float* ab = a + (size_t)b*NN;
  float mx = -1e30f;
  for (int i=t;i<NN;i+=1024) mx = fmaxf(mx, ab[i]);
  red[t]=mx; __syncthreads();
  for (int o=512;o;o>>=1){ if(t<o) red[t]=fmaxf(red[t],red[t+o]); __syncthreads(); }
  mx = red[0]; __syncthreads();
  float sm=0.f;
  for (int i=t;i<NN;i+=1024) sm += __expf(ab[i]-mx);
  red[t]=sm; __syncthreads();
  for (int o=512;o;o>>=1){ if(t<o) red[t]+=red[t+o]; __syncthreads(); }
  float inv = (float)SS/red[0];
  for (int i=t;i<NN;i+=1024){
    float av = __expf(ab[i]-mx)*inv;
    log_a[(size_t)b*NN+i] = __logf(av + EPS8);
  }
}

// ---------------- WMMA GEMM: k = x@Wk^T, v = x@Wv^T (f16 in, f32 out) ----------------
__global__ __launch_bounds__(256) void gemm_kv(const _Float16* __restrict__ x16,
    const float* __restrict__ Wk, const float* __restrict__ Wv,
    float* __restrict__ kOut, float* __restrict__ vOut){
  // Pre-pack B fragments (32x16 f16, CDNA5 B layout) for both matrices.
  __shared__ __align__(32) _Float16 bfrag[16][32][16];   // 16 KB
  int tid = threadIdx.x;
  for (int idx = tid; idx < 16*32*16; idx += 256){
    int f = idx >> 9; int lane = (idx >> 4) & 31; int e = idx & 15;
    int mat = f >> 3, ks = (f >> 2) & 1, nt = f & 3;
    int hi = lane >> 4, nl = lane & 15;
    int K = ks*32 + (e < 8 ? hi*8 + e : 16 + hi*8 + (e - 8));
    int n = nt*16 + nl;
    const float* W = mat ? Wv : Wk;
    bfrag[f][lane][e] = (_Float16)W[n*64 + K];   // B[K][n] = W[n][K]
  }
  __syncthreads();
  int wave = tid >> 5, lane = tid & 31;
  size_t row0 = (size_t)blockIdx.x*128 + wave*16;
  int m = lane & 15, hi = lane >> 4;
  const _Float16* p = x16 + (row0 + m)*64;
  // A fragments per CDNA5 16-bit A layout: two 16B loads per kstep
  v16h a0, a1;
  {
    v8h l0 = *(const v8h*)(p + hi*8);
    v8h h0 = *(const v8h*)(p + 16 + hi*8);
    v8h l1 = *(const v8h*)(p + 32 + hi*8);
    v8h h1 = *(const v8h*)(p + 48 + hi*8);
    #pragma unroll
    for (int i=0;i<8;i++){ a0[i]=l0[i]; a0[8+i]=h0[i]; a1[i]=l1[i]; a1[8+i]=h1[i]; }
  }
  v8f acc[8];
  #pragma unroll
  for (int t=0;t<8;t++){ acc[t] = (v8f)(0.f); }
  #pragma unroll
  for (int mat=0; mat<2; mat++){
    #pragma unroll
    for (int nt=0; nt<4; nt++){
      int t = mat*4+nt;
      v16h b0 = *(const v16h*)&bfrag[mat*8 + nt][lane][0];
      v16h b1 = *(const v16h*)&bfrag[mat*8 + 4 + nt][lane][0];
      acc[t] = __builtin_amdgcn_wmma_f32_16x16x32_f16(false, a0, false, b0, (short)0, acc[t], false, false);
      acc[t] = __builtin_amdgcn_wmma_f32_16x16x32_f16(false, a1, false, b1, (short)0, acc[t], false, false);
    }
  }
  int nl = lane & 15; int mb = (lane >> 4)*8;
  #pragma unroll
  for (int mat=0; mat<2; mat++){
    float* O = mat ? vOut : kOut;
    #pragma unroll
    for (int nt=0; nt<4; nt++){
      v8f c = acc[mat*4+nt];
      #pragma unroll
      for (int r=0;r<8;r++)
        O[(row0 + mb + r)*64 + nt*16 + nl] = c[r];
    }
  }
}

// ---------------- per-iteration slot prep: LN, q, b-marginal ----------------
__global__ __launch_bounds__(64) void slot_prep(const float* __restrict__ slots,
    const float* __restrict__ lng, const float* __restrict__ lnb,
    const float* __restrict__ Wq, const float* __restrict__ ws_w, const float* __restrict__ ws_b,
    float* __restrict__ q, float* __restrict__ qn, float* __restrict__ log_b){
  __shared__ float sln[64], red[64], bl[SS];
  int b = blockIdx.x, t = threadIdx.x;
  for (int s=0;s<SS;s++){
    float x = slots[((size_t)b*SS+s)*64+t];
    red[t]=x; __syncthreads();
    for (int o=32;o;o>>=1){ if(t<o) red[t]+=red[t+o]; __syncthreads(); }
    float mean = red[0]*(1.f/64.f); __syncthreads();
    float d = x-mean; red[t]=d*d; __syncthreads();
    for (int o=32;o;o>>=1){ if(t<o) red[t]+=red[t+o]; __syncthreads(); }
    float rstd = rsqrtf(red[0]*(1.f/64.f)+LNE); __syncthreads();
    float y = d*rstd*lng[t]+lnb[t];
    sln[t]=y;
    red[t]=y*ws_w[t]; __syncthreads();
    for (int o=32;o;o>>=1){ if(t<o) red[t]+=red[t+o]; __syncthreads(); }
    if (t==0) bl[s]=red[0]+ws_b[0];
    __syncthreads();
    float qd=0.f;
    #pragma unroll
    for (int c=0;c<64;c++) qd += sln[c]*Wq[t*64+c];
    q[((size_t)b*SS+s)*64+t]=qd;
    red[t]=qd*qd; __syncthreads();
    for (int o=32;o;o>>=1){ if(t<o) red[t]+=red[t+o]; __syncthreads(); }
    if (t==0) qn[b*SS+s]=red[0];
    __syncthreads();
  }
  if (t==0){
    float mx=-1e30f;
    for (int s=0;s<SS;s++) mx=fmaxf(mx,bl[s]);
    float sm=0.f;
    for (int s=0;s<SS;s++) sm+=__expf(bl[s]-mx);
    for (int s=0;s<SS;s++){
      float bm = __expf(bl[s]-mx)/sm*(float)SS;
      log_b[b*SS+s] = __logf(bm + EPS8);
    }
  }
}

// ---------------- cdist(k,q) -> C [B,S,N] ----------------
__global__ __launch_bounds__(128) void cdist_kernel(const float* __restrict__ k,
    const float* __restrict__ q, const float* __restrict__ qn, float* __restrict__ C){
  __shared__ float kl[128][65];
  __shared__ float qs[SS][64];
  __shared__ float qns[SS];
  int t = threadIdx.x;
  size_t row0 = (size_t)blockIdx.x*128;
  int b = (int)(row0 >> 14);
  for (int i=t;i<128*64;i+=128){ int r=i>>6, c=i&63; kl[r][c]=k[(row0+r)*64+c]; }
  for (int i=t;i<SS*64;i+=128){ qs[i>>6][i&63]=q[(size_t)b*SS*64+i]; }
  if (t<SS) qns[t]=qn[b*SS+t];
  __syncthreads();
  float kn=0.f;
  #pragma unroll
  for (int c=0;c<64;c++){ float x=kl[t][c]; kn += x*x; }
  int n = (int)(row0 & (NN-1)) + t;
  for (int s=0;s<SS;s++){
    float dot=0.f;
    #pragma unroll
    for (int c=0;c<64;c++) dot += kl[t][c]*qs[s][c];
    float sq = kn + qns[s] - 2.f*dot;
    C[((size_t)b*SS+s)*NN + n] = sqrtf(fmaxf(sq,1e-12f));
  }
}

// ---------------- Sinkhorn u-update: u = log_a - LSE_s(-C + v) ----------------
__global__ __launch_bounds__(256) void sink_u(const float* __restrict__ C,
    const float* __restrict__ v, const float* __restrict__ log_a, float* __restrict__ u){
  size_t mI = (size_t)blockIdx.x*256 + threadIdx.x;
  int b = (int)(mI >> 14); size_t n = mI & (NN-1);
  const float* Cb = C + (size_t)b*SS*NN + n;
  const float* vb = v + b*SS;
  float kv[SS]; float mx=-1e30f;
  #pragma unroll
  for (int s=0;s<SS;s++){ kv[s] = vb[s] - Cb[(size_t)s*NN]; mx = fmaxf(mx, kv[s]); }
  float sm=0.f;
  #pragma unroll
  for (int s=0;s<SS;s++) sm += __expf(kv[s]-mx);
  u[mI] = log_a[mI] - (mx + __logf(sm));
}

// ---------------- Sinkhorn v-update: v = log_b - LSE_n(-C + u) ----------------
__global__ __launch_bounds__(1024) void sink_v(const float* __restrict__ C,
    const float* __restrict__ u, const float* __restrict__ log_b, float* __restrict__ v){
  __shared__ float rm[1024], rs[1024];
  int bs = blockIdx.x, b = bs/SS, t = threadIdx.x;
  const float* Cr = C + (size_t)bs*NN;
  const float* ub = u + (size_t)b*NN;
  float mx=-1e30f, sm=0.f;
  for (int n=t;n<NN;n+=1024){
    float kv = ub[n]-Cr[n];
    if (kv>mx){ sm = sm*__expf(mx-kv)+1.f; mx=kv; } else sm += __expf(kv-mx);
  }
  rm[t]=mx; rs[t]=sm; __syncthreads();
  for (int o=512;o;o>>=1){
    if (t<o){
      float m1=rm[t], s1=rs[t], m2=rm[t+o], s2=rs[t+o];
      float mN=fmaxf(m1,m2);
      rs[t]=s1*__expf(m1-mN)+s2*__expf(m2-mN); rm[t]=mN;
    }
    __syncthreads();
  }
  if (t==0) v[bs] = log_b[bs] - (rm[0] + __logf(rs[0]));
}

// ---------------- backward init: Kbar = gP*P, ubar_T row-sums ----------------
__global__ __launch_bounds__(256) void bwd_init(const float* __restrict__ C,
    const float* __restrict__ uT, const float* __restrict__ vT,
    float* __restrict__ Kbar, float* __restrict__ ubar){
  size_t mI = (size_t)blockIdx.x*256 + threadIdx.x;
  int b = (int)(mI >> 14); size_t n = mI & (NN-1);
  float uu = uT[mI]; float acc=0.f;
  #pragma unroll
  for (int s=0;s<SS;s++){
    size_t idx = ((size_t)b*SS+s)*NN + n;
    float P = __expf(uu + vT[b*SS+s] - C[idx]);
    float gP = -(__logf(P+EPS8) + P/(P+EPS8));
    float kb = gP*P;
    Kbar[idx]=kb; acc+=kb;
  }
  ubar[mI]=acc;
}

// ---------------- column sum of Kbar rows -> vbar ----------------
__global__ __launch_bounds__(1024) void colsum(const float* __restrict__ Kbar, float* __restrict__ vbar){
  __shared__ float red[1024];
  int bs=blockIdx.x, t=threadIdx.x;
  const float* r = Kbar + (size_t)bs*NN;
  float a=0.f;
  for (int n=t;n<NN;n+=1024) a += r[n];
  red[t]=a; __syncthreads();
  for (int o=512;o;o>>=1){ if(t<o) red[t]+=red[t+o]; __syncthreads(); }
  if (t==0) vbar[bs]=red[0];
}

// ---------------- backward through v_t: Kbar -= vbar*p ; ubar = init - sum_s vbar*p ----------------
__global__ __launch_bounds__(256) void bwd_v(const float* __restrict__ C,
    const float* __restrict__ u_t, const float* __restrict__ v_t,
    const float* __restrict__ vbar, const float* __restrict__ log_b,
    float* __restrict__ Kbar, float* __restrict__ ubar, int addInit){
  size_t mI = (size_t)blockIdx.x*256 + threadIdx.x;
  int b = (int)(mI >> 14); size_t n = mI & (NN-1);
  float uu = u_t[mI]; float acc=0.f;
  #pragma unroll
  for (int s=0;s<SS;s++){
    size_t idx = ((size_t)b*SS+s)*NN + n;
    float p = __expf(uu - C[idx] - log_b[b*SS+s] + v_t[b*SS+s]);
    float vb = vbar[b*SS+s];
    Kbar[idx] -= vb*p;
    acc -= vb*p;
  }
  ubar[mI] = (addInit ? ubar[mI] : 0.f) + acc;
}

// ---------------- backward through u_t (reduce): vbar_prev = -sum_n ubar*q ----------------
__global__ __launch_bounds__(1024) void bwd_u_reduce(const float* __restrict__ C,
    const float* __restrict__ ubar, const float* __restrict__ u_t,
    const float* __restrict__ v_prev, const float* __restrict__ log_a,
    float* __restrict__ vbar_out){
  __shared__ float red[1024];
  int bs=blockIdx.x, b=bs/SS, t=threadIdx.x;
  const float* Cr = C + (size_t)bs*NN;
  const float* ub = ubar + (size_t)b*NN;
  const float* ut = u_t + (size_t)b*NN;
  const float* la = log_a + (size_t)b*NN;
  float vp = v_prev[bs];
  float acc=0.f;
  for (int n=t;n<NN;n+=1024){
    float qv = __expf(vp - Cr[n] - la[n] + ut[n]);
    acc -= ub[n]*qv;
  }
  red[t]=acc; __syncthreads();
  for (int o=512;o;o>>=1){ if(t<o) red[t]+=red[t+o]; __syncthreads(); }
  if (t==0) vbar_out[bs]=red[0];
}

// ---------------- backward through u_t (elementwise): Kbar -= ubar*q ----------------
__global__ __launch_bounds__(256) void bwd_u_elem(const float* __restrict__ C,
    const float* __restrict__ ubar, const float* __restrict__ u_t,
    const float* __restrict__ v_prev, const float* __restrict__ log_a,
    float* __restrict__ Kbar){
  size_t mI = (size_t)blockIdx.x*256 + threadIdx.x;
  int b = (int)(mI >> 14); size_t n = mI & (NN-1);
  float uu = u_t[mI], la = log_a[mI], ub = ubar[mI];
  #pragma unroll
  for (int s=0;s<SS;s++){
    size_t idx = ((size_t)b*SS+s)*NN + n;
    float qv = __expf(v_prev[b*SS+s] - C[idx] - la + uu);
    Kbar[idx] -= ub*qv;
  }
}

// ---------------- per-batch grad norm ----------------
__global__ __launch_bounds__(1024) void gradnorm(const float* __restrict__ Kbar, float* __restrict__ gn){
  __shared__ float red[1024];
  int b=blockIdx.x, t=threadIdx.x;
  const float* p = Kbar + (size_t)b*SNP;
  float a=0.f;
  for (int i=t;i<SNP;i+=1024){ float x=p[i]; a+=x*x; }
  red[t]=a; __syncthreads();
  for (int o=512;o;o>>=1){ if(t<o) red[t]+=red[t+o]; __syncthreads(); }
  if (t==0) gn[b]=sqrtf(red[0])+EPS8;
}

// ---------------- MESH cost update: C += lr*Kbar/gn (dE/dC = -Kbar) ----------------
__global__ __launch_bounds__(256) void mesh_update(float* __restrict__ C,
    const float* __restrict__ Kbar, const float* __restrict__ gn){
  size_t i = (size_t)blockIdx.x*256 + threadIdx.x;
  int b = (int)(i / SNP);
  C[i] += Kbar[i] / gn[b];
}

// ---------------- attention: P, slot_pos, updates = P@v ----------------
__global__ __launch_bounds__(256) void attn_update(const float* __restrict__ C,
    const float* __restrict__ u, const float* __restrict__ v,
    const float* __restrict__ vfeat, float* __restrict__ attn_out,
    float* __restrict__ pos_out, float* __restrict__ updates){
  __shared__ float Ps[256];
  __shared__ float upd[4][64];
  __shared__ float red[256];
  int bs=blockIdx.x, b=bs/SS, t=threadIdx.x;
  int grp=t>>6, d=t&63;
  const float* Cr = C + (size_t)bs*NN;
  const float* ub = u + (size_t)b*NN;
  const float* Vb = vfeat + (size_t)b*NN*64;
  float vv = v[bs];
  float acc=0.f, px=0.f, py=0.f;
  for (int n0=0;n0<NN;n0+=256){
    int n = n0 + t;
    float P = __expf(ub[n] + vv - Cr[n]);
    attn_out[(size_t)bs*NN + n] = P;
    px += P*(float)(n & 127);
    py += P*(float)(n >> 7);
    __syncthreads();
    Ps[t]=P; __syncthreads();
    if (n0+256 < NN) __builtin_prefetch(Vb + (size_t)(n0+256+grp)*64 + d, 0, 1);
    for (int j=grp;j<256;j+=4) acc += Ps[j]*Vb[(size_t)(n0+j)*64 + d];
  }
  upd[grp][d]=acc;
  red[t]=px; __syncthreads();
  for (int o=128;o;o>>=1){ if(t<o) red[t]+=red[t+o]; __syncthreads(); }
  if (t==0) pos_out[(size_t)bs*2+0]=red[0]*(1.f/127.f);
  __syncthreads();
  red[t]=py; __syncthreads();
  for (int o=128;o;o>>=1){ if(t<o) red[t]+=red[t+o]; __syncthreads(); }
  if (t==0) pos_out[(size_t)bs*2+1]=red[0]*(1.f/127.f);
  __syncthreads();
  if (t<64) updates[(size_t)bs*64+t]=upd[0][t]+upd[1][t]+upd[2][t]+upd[3][t];
}

// ---------------- GRU cell + FF residual ----------------
__global__ __launch_bounds__(192) void gru_ff(const float* __restrict__ updates,
    const float* __restrict__ slots_prev,
    const float* __restrict__ W_ih, const float* __restrict__ W_hh,
    const float* __restrict__ b_ih, const float* __restrict__ b_hh,
    const float* __restrict__ fc1_w, const float* __restrict__ fc1_b,
    const float* __restrict__ fc2_w, const float* __restrict__ fc2_b,
    const float* __restrict__ lng, const float* __restrict__ lnb,
    float* __restrict__ slots_out, float* __restrict__ out_slots){
  __shared__ float xs[64], hs[64], gi[192], gh[192], hn[64], lnh[64], ff1[128], red[64];
  int bs=blockIdx.x, t=threadIdx.x;
  if (t<64){ xs[t]=updates[(size_t)bs*64+t]; hs[t]=slots_prev[(size_t)bs*64+t]; }
  __syncthreads();
  float a1=b_ih[t], a2=b_hh[t];
  for (int c=0;c<64;c++){ a1 += xs[c]*W_ih[t*64+c]; a2 += hs[c]*W_hh[t*64+c]; }
  gi[t]=a1; gh[t]=a2; __syncthreads();
  if (t<64){
    float r = 1.f/(1.f+__expf(-(gi[t]+gh[t])));
    float z = 1.f/(1.f+__expf(-(gi[64+t]+gh[64+t])));
    float nn2 = tanhf(gi[128+t] + r*gh[128+t]);
    hn[t] = (1.f-z)*nn2 + z*hs[t];
  }
  __syncthreads();
  if (t<64) red[t]=hn[t];
  __syncthreads();
  for (int o=32;o;o>>=1){ if(t<o) red[t]+=red[t+o]; __syncthreads(); }
  float mean = red[0]*(1.f/64.f); __syncthreads();
  if (t<64){ float d=hn[t]-mean; red[t]=d*d; }
  __syncthreads();
  for (int o=32;o;o>>=1){ if(t<o) red[t]+=red[t+o]; __syncthreads(); }
  float rstd = rsqrtf(red[0]*(1.f/64.f)+LNE); __syncthreads();
  if (t<64) lnh[t] = (hn[t]-mean)*rstd*lng[t]+lnb[t];
  __syncthreads();
  if (t<128){
    float s=fc1_b[t];
    for (int c=0;c<64;c++) s += lnh[c]*fc1_w[t*64+c];
    ff1[t]=fmaxf(s,0.f);
  }
  __syncthreads();
  if (t<64){
    float s=fc2_b[t];
    for (int c=0;c<128;c++) s += ff1[c]*fc2_w[t*128+c];
    float o_ = hn[t]+s;
    slots_out[(size_t)bs*64+t]=o_;
    out_slots[(size_t)bs*64+t]=o_;
  }
}

// =====================================================================
extern "C" void kernel_launch(void* const* d_in, const int* in_sizes, int n_in,
                              void* d_out, int out_size, void* d_ws, size_t ws_size,
                              hipStream_t stream){
  const float* inputs   = (const float*)d_in[0];
  const float* noise    = (const float*)d_in[1];
  const float* slots_mu = (const float*)d_in[2];
  const float* slots_sg = (const float*)d_in[3];
  const float* Wq  = (const float*)d_in[4];
  const float* Wk  = (const float*)d_in[5];
  const float* Wv  = (const float*)d_in[6];
  const float* W_ih= (const float*)d_in[7];
  const float* W_hh= (const float*)d_in[8];
  const float* b_ih= (const float*)d_in[9];
  const float* b_hh= (const float*)d_in[10];
  const float* fc1_w=(const float*)d_in[11];
  const float* fc1_b=(const float*)d_in[12];
  const float* fc2_w=(const float*)d_in[13];
  const float* fc2_b=(const float*)d_in[14];
  const float* ln_in_g=(const float*)d_in[15];
  const float* ln_in_b=(const float*)d_in[16];
  const float* ln_sl_g=(const float*)d_in[17];
  const float* ln_sl_b=(const float*)d_in[18];
  const float* ln_ff_g=(const float*)d_in[19];
  const float* ln_ff_b=(const float*)d_in[20];
  const float* wi_w=(const float*)d_in[21];
  const float* wi_b=(const float*)d_in[22];
  const float* ws_w=(const float*)d_in[23];
  const float* ws_b=(const float*)d_in[24];
  (void)in_sizes; (void)n_in; (void)out_size; (void)ws_size;

  char* w = (char*)d_ws;
  auto alloc = [&](size_t bytes)->void*{
    void* p = (void*)w;
    w += (bytes + 255) & ~(size_t)255;
    return p;
  };
  _Float16* x16   = (_Float16*)alloc((size_t)MM*64*2);
  float* kbuf     = (float*)alloc((size_t)MM*64*4);
  float* vbuf     = (float*)alloc((size_t)MM*64*4);
  float* aLog     = (float*)alloc((size_t)MM*4);
  float* log_a    = (float*)alloc((size_t)MM*4);
  float* Cbuf     = (float*)alloc((size_t)BSN*4);
  float* Kbar     = (float*)alloc((size_t)BSN*4);
  float* u_hist   = (float*)alloc((size_t)9*MM*4);
  float* u_cur    = (float*)alloc((size_t)MM*4);
  float* ubar     = (float*)alloc((size_t)MM*4);
  float* v_hist   = (float*)alloc((size_t)9*BS*4);
  float* v_cur    = (float*)alloc((size_t)BS*4);
  float* vbarA    = (float*)alloc((size_t)BS*4);
  float* vbarB    = (float*)alloc((size_t)BS*4);
  float* qbuf     = (float*)alloc((size_t)BS*64*4);
  float* qn       = (float*)alloc((size_t)BS*4);
  float* log_b    = (float*)alloc((size_t)BS*4);
  float* slotsA   = (float*)alloc((size_t)BS*64*4);
  float* slotsB   = (float*)alloc((size_t)BS*64*4);
  float* updates  = (float*)alloc((size_t)BS*64*4);
  float* gn       = (float*)alloc((size_t)BB*4);

  float* out        = (float*)d_out;
  float* out_slots  = out;                 // [B,S,D]
  float* out_pos    = out + BS*DD;         // [B,S,2]
  float* out_attn   = out + BS*DD + BS*2;  // [B,S,N]

  // v_hist[0] stays zero for every mesh forward pass
  hipMemsetAsync(v_hist, 0, (size_t)BS*4, stream);

  slots_init<<<(BS*DD+255)/256, 256, 0, stream>>>(slots_mu, slots_sg, noise, slotsA);
  ln_a_kernel<<<MM/8, 256, 0, stream>>>(inputs, ln_in_g, ln_in_b, wi_w, wi_b, x16, aLog);
  a_softmax<<<BB, 1024, 0, stream>>>(aLog, log_a);
  gemm_kv<<<MM/128, 256, 0, stream>>>(x16, Wk, Wv, kbuf, vbuf);

  float* scur = slotsA;
  float* snext = slotsB;
  for (int it=0; it<3; it++){
    slot_prep<<<BB, 64, 0, stream>>>(scur, ln_sl_g, ln_sl_b, Wq, ws_w, ws_b, qbuf, qn, log_b);
    cdist_kernel<<<MM/128, 128, 0, stream>>>(kbuf, qbuf, qn, Cbuf);

    for (int mi=0; mi<4; mi++){
      // forward Sinkhorn (8 steps), keep history
      for (int tt=1; tt<=8; tt++){
        sink_u<<<MM/256, 256, 0, stream>>>(Cbuf, v_hist+(size_t)(tt-1)*BS, log_a, u_hist+(size_t)tt*MM);
        sink_v<<<BS, 1024, 0, stream>>>(Cbuf, u_hist+(size_t)tt*MM, log_b, v_hist+(size_t)tt*BS);
      }
      // entropy gradient backward
      bwd_init<<<MM/256, 256, 0, stream>>>(Cbuf, u_hist+(size_t)8*MM, v_hist+(size_t)8*BS, Kbar, ubar);
      colsum<<<BS, 1024, 0, stream>>>(Kbar, vbarA);
      float* vb_cur = vbarA; float* vb_prev = vbarB;
      for (int tt=8; tt>=1; tt--){
        bwd_v<<<MM/256, 256, 0, stream>>>(Cbuf, u_hist+(size_t)tt*MM, v_hist+(size_t)tt*BS,
                                          vb_cur, log_b, Kbar, ubar, tt==8 ? 1 : 0);
        bwd_u_reduce<<<BS, 1024, 0, stream>>>(Cbuf, ubar, u_hist+(size_t)tt*MM,
                                              v_hist+(size_t)(tt-1)*BS, log_a, vb_prev);
        bwd_u_elem<<<MM/256, 256, 0, stream>>>(Cbuf, ubar, u_hist+(size_t)tt*MM,
                                               v_hist+(size_t)(tt-1)*BS, log_a, Kbar);
        float* tmp = vb_cur; vb_cur = vb_prev; vb_prev = tmp;
      }
      gradnorm<<<BB, 1024, 0, stream>>>(Kbar, gn);
      mesh_update<<<BSN/256, 256, 0, stream>>>(Cbuf, Kbar, gn);
    }

    // mesh's final sinkhorn (8 from zero) + warm-started attention sinkhorn (8 more) == 16 from zero
    hipMemsetAsync(v_cur, 0, (size_t)BS*4, stream);
    for (int tt=0; tt<16; tt++){
      sink_u<<<MM/256, 256, 0, stream>>>(Cbuf, v_cur, log_a, u_cur);
      sink_v<<<BS, 1024, 0, stream>>>(Cbuf, u_cur, log_b, v_cur);
    }

    attn_update<<<BS, 256, 0, stream>>>(Cbuf, u_cur, v_cur, vbuf, out_attn, out_pos, updates);
    gru_ff<<<BS, 192, 0, stream>>>(updates, scur, W_ih, W_hh, b_ih, b_hh,
                                   fc1_w, fc1_b, fc2_w, fc2_b, ln_ff_g, ln_ff_b,
                                   snext, out_slots);
    float* tmp = scur; scur = snext; snext = tmp;
  }
}
